// SO3_Linear2Scalar_e2former_89979564851592
// MI455X (gfx1250) — matrix operand order repaired and checked by reading, not compiled, not run on hardware
//
#include <hip/hip_runtime.h>

typedef __attribute__((ext_vector_type(16))) __bf16 v16bf;
typedef __attribute__((ext_vector_type(8)))  __bf16 v8bf;
typedef __attribute__((ext_vector_type(8)))  float  v8f;

#define LMAX   4
#define IN_F   128
#define OUT_F  128
#define L_SUM  25
#define HALF   64
#define FEAT   320            // HALF*(LMAX+1)

// d_ws layout, offsets in __bf16 elements
#define PW_OFF    0           // packed weight   : 5*64*128 = 40960
#define PW2_OFF   40960       // packed weight2  : 40960
#define PW1_OFF   81920       // packed w1       : 320*128 = 40960
#define PW2M_OFF  122880      // packed w2       : 128*128 = 16384
#define FEAT_OFF  139264      // feat bf16       : B*320
// total ws ~ 42.2 MB

// ---------------------------------------------------------------------------
// Pack an f32 matrix set into CDNA5 WMMA bf16 B-fragment order.
// Element (l, k, n) -> fragment ((l*NT + n/16)*KT + k/32), lane = (n&15) + (((k&31)>>3)&1)*16,
// pos = (k&7) + ((k&31)>>4)*8.  Each fragment = 512 bf16 = 1 KB, lane-major (16 contiguous bf16 per lane).
// ---------------------------------------------------------------------------
__global__ __launch_bounds__(256) void pack_b_frag(const float* __restrict__ src,
                                                   __bf16* __restrict__ dst,
                                                   int L, int Kdim, int Ndim,
                                                   int ls, int ks, int ns)
{
    int idx = blockIdx.x * 256 + threadIdx.x;
    int total = L * Kdim * Ndim;
    if (idx >= total) return;
    int n = idx % Ndim;
    int k = (idx / Ndim) % Kdim;
    int l = idx / (Ndim * Kdim);
    float v = src[(size_t)l * ls + (size_t)k * ks + (size_t)n * ns];
    int KT = Kdim >> 5, NT = Ndim >> 4;
    int kt = k >> 5, kl = k & 31;
    int nt = n >> 4, nl = n & 15;
    int lane = nl + ((kl >> 3) & 1) * 16;
    int pos  = (kl & 7) + (kl >> 4) * 8;
    int frag = (l * NT + nt) * KT + kt;
    dst[(size_t)frag * 512 + lane * 16 + pos] = (__bf16)v;
}

// ---------------------------------------------------------------------------
// Stage 1: per-degree dual linear + gated product + segment-sum -> feat[B,320] (bf16)
// Block: 256 threads (8 waves). Wave = 16 batch rows x 32 output cols.
// Block covers 64 rows x 64 cols. Packed W|W2 (160 KB bf16) staged in LDS.
// ---------------------------------------------------------------------------
__global__ __launch_bounds__(256) void so3_stage1(const float* __restrict__ inp,
                                                  const float* __restrict__ bias,
                                                  const __bf16* __restrict__ pwAll,
                                                  __bf16* __restrict__ featg)
{
    extern __shared__ char lds[];                 // 163840 bytes: pw | pw2
    {
        const uint4* s = (const uint4*)pwAll;
        uint4* d = (uint4*)lds;
        for (int i = threadIdx.x; i < 163840 / 16; i += 256) d[i] = s[i];
    }
    __syncthreads();

    const int tid      = threadIdx.x;
    const int lane     = tid & 31;
    const int wv       = tid >> 5;
    const int colGroup = wv & 1;                  // 0..1 -> 32-col half of HALF=64
    const int rowGroup = wv >> 1;                 // 0..3 -> 16-row group
    const int rowBase  = blockIdx.x * 64 + rowGroup * 16;
    const int nl       = lane & 15;
    const int g        = lane >> 4;
    const int aRow     = rowBase + nl;            // A-fragment source row for this lane
    const int cRow     = rowBase + g * 8;         // C/D row base for this lane

    float bv[2];
    bv[0] = bias[colGroup * 32 + nl];
    bv[1] = bias[colGroup * 32 + 16 + nl];

    v8f acc[LMAX + 1][2] = {};                    // segment-sum accumulators (feat)

    for (int l = 0; l <= LMAX; ++l) {
        const int mEnd = (l + 1) * (l + 1);
        for (int mm = l * l; mm < mEnd; ++mm) {
            v8f s1[2] = {};
            v8f s2[2] = {};
            const float* arow = inp + ((size_t)aRow * L_SUM + mm) * IN_F;
            if (mm + 1 < L_SUM) __builtin_prefetch(arow + IN_F, 0, 0);

#pragma unroll
            for (int kt = 0; kt < 4; ++kt) {
                // Build bf16 A fragment (16x32) in ISA layout from f32 global
                const float* ap = arow + kt * 32 + g * 8;
                float4 f0 = *(const float4*)(ap);
                float4 f1 = *(const float4*)(ap + 4);
                float4 f2 = *(const float4*)(ap + 16);
                float4 f3 = *(const float4*)(ap + 20);
                v16bf a;
                a[0]  = (__bf16)f0.x; a[1]  = (__bf16)f0.y; a[2]  = (__bf16)f0.z; a[3]  = (__bf16)f0.w;
                a[4]  = (__bf16)f1.x; a[5]  = (__bf16)f1.y; a[6]  = (__bf16)f1.z; a[7]  = (__bf16)f1.w;
                a[8]  = (__bf16)f2.x; a[9]  = (__bf16)f2.y; a[10] = (__bf16)f2.z; a[11] = (__bf16)f2.w;
                a[12] = (__bf16)f3.x; a[13] = (__bf16)f3.y; a[14] = (__bf16)f3.z; a[15] = (__bf16)f3.w;

#pragma unroll
                for (int nt = 0; nt < 2; ++nt) {
                    const int ntg  = colGroup * 2 + nt;            // 0..3
                    const int frag = (l * 4 + ntg) * 4 + kt;
                    const v16bf bw  = *(const v16bf*)(lds + (size_t)frag * 1024 + lane * 32);
                    const v16bf bw2 = *(const v16bf*)(lds + 81920 + (size_t)frag * 1024 + lane * 32);
                    s1[nt] = __builtin_amdgcn_wmma_f32_16x16x32_bf16(false, a, false, bw,  (short)0, s1[nt], false, false);
                    s2[nt] = __builtin_amdgcn_wmma_f32_16x16x32_bf16(false, a, false, bw2, (short)0, s2[nt], false, false);
                }
            }

            if (mm == 0) {                         // bias only on the l=0 row
#pragma unroll
                for (int nt = 0; nt < 2; ++nt)
#pragma unroll
                    for (int r = 0; r < 8; ++r) s1[nt][r] += bv[nt];
            }
#pragma unroll
            for (int nt = 0; nt < 2; ++nt)
#pragma unroll
                for (int r = 0; r < 8; ++r)
                    acc[l][nt][r] = __builtin_fmaf(s1[nt][r], s2[nt][r], acc[l][nt][r]);
        }
    }

    // Store feat (bf16, row-major [B,320])
#pragma unroll
    for (int l = 0; l <= LMAX; ++l)
#pragma unroll
        for (int nt = 0; nt < 2; ++nt) {
            const int col = l * HALF + colGroup * 32 + nt * 16 + nl;
#pragma unroll
            for (int r = 0; r < 8; ++r)
                featg[(size_t)(cRow + r) * FEAT + col] = (__bf16)acc[l][nt][r];
        }
}

// ---------------------------------------------------------------------------
// Stage 2: feat@w1 -> LayerNorm -> SiLU -> @w2. Block = 32 rows, 8 waves
// (wave = 16 rows x 32 cols). w1/w2 packed bf16 staged in LDS (112 KB) +
// h f32 (16 KB) + h_bf16 (8 KB).
// ---------------------------------------------------------------------------
__global__ __launch_bounds__(256) void so3_stage2(const __bf16* __restrict__ featg,
                                                  const __bf16* __restrict__ pmats, // pw1|pw2m
                                                  const float* __restrict__ b1,
                                                  const float* __restrict__ gamma,
                                                  const float* __restrict__ beta,
                                                  const float* __restrict__ b2,
                                                  float* __restrict__ outp)
{
    extern __shared__ char lds[];                 // 139264 bytes
    {
        const uint4* s = (const uint4*)pmats;
        uint4* d = (uint4*)lds;
        for (int i = threadIdx.x; i < (81920 + 32768) / 16; i += 256) d[i] = s[i];
    }
    __syncthreads();
    float*  hf = (float*)(lds + 114688);          // 32 x 128 f32
    __bf16* hb = (__bf16*)(lds + 131072);         // 32 x 128 bf16

    const int tid      = threadIdx.x;
    const int lane     = tid & 31;
    const int wv       = tid >> 5;
    const int colGroup = wv & 3;                  // 0..3 -> 32 of 128 cols
    const int rowGroup = wv >> 2;                 // 0..1 -> 16 rows
    const int nl       = lane & 15;
    const int g        = lane >> 4;
    const int rowBase  = blockIdx.x * 32 + rowGroup * 16;

    // ---- matmul 1: feat[16x320] @ w1[320x128] ----
    v8f c[2] = {};
    const __bf16* fr = featg + (size_t)(rowBase + nl) * FEAT;
#pragma unroll
    for (int kt = 0; kt < 10; ++kt) {
        v8bf lo = *(const v8bf*)(fr + kt * 32 + g * 8);
        v8bf hi = *(const v8bf*)(fr + kt * 32 + g * 8 + 16);
        v16bf a = __builtin_shufflevector(lo, hi, 0,1,2,3,4,5,6,7,8,9,10,11,12,13,14,15);
#pragma unroll
        for (int nt = 0; nt < 2; ++nt) {
            const int frag = (colGroup * 2 + nt) * 10 + kt;
            const v16bf bw = *(const v16bf*)(lds + (size_t)frag * 1024 + lane * 32);
            c[nt] = __builtin_amdgcn_wmma_f32_16x16x32_bf16(false, a, false, bw, (short)0, c[nt], false, false);
        }
    }
#pragma unroll
    for (int nt = 0; nt < 2; ++nt) {
        const int col = colGroup * 32 + nt * 16 + nl;
        const float bb = b1[col];
#pragma unroll
        for (int r = 0; r < 8; ++r)
            hf[(rowGroup * 16 + g * 8 + r) * 128 + col] = c[nt][r] + bb;
    }
    __syncthreads();

    // ---- LayerNorm + SiLU over 32 rows x 128 cols (8 lanes per row) ----
    {
        const int row = tid >> 3;                 // 0..31
        const int sub = tid & 7;
        const float* hr = hf + row * 128 + sub * 16;
        float4 q0 = *(const float4*)(hr);
        float4 q1 = *(const float4*)(hr + 4);
        float4 q2 = *(const float4*)(hr + 8);
        float4 q3 = *(const float4*)(hr + 12);
        float v[16] = { q0.x,q0.y,q0.z,q0.w, q1.x,q1.y,q1.z,q1.w,
                        q2.x,q2.y,q2.z,q2.w, q3.x,q3.y,q3.z,q3.w };
        float s = 0.f, s2 = 0.f;
#pragma unroll
        for (int j = 0; j < 16; ++j) { s += v[j]; s2 = __builtin_fmaf(v[j], v[j], s2); }
#pragma unroll
        for (int m = 1; m < 8; m <<= 1) { s += __shfl_xor(s, m, 32); s2 += __shfl_xor(s2, m, 32); }
        const float mean = s * (1.0f / 128.0f);
        const float var  = s2 * (1.0f / 128.0f) - mean * mean;
        const float inv  = rsqrtf(var + 1e-5f);
#pragma unroll
        for (int j = 0; j < 16; ++j) {
            const int col = sub * 16 + j;
            float y = (v[j] - mean) * inv * gamma[col] + beta[col];
            float si = y / (1.0f + __expf(-y));   // SiLU
            hb[row * 128 + col] = (__bf16)si;
        }
    }
    __syncthreads();

    // ---- matmul 2: hn[16x128] @ w2[128x128] ----
    v8f d[2] = {};
    const __bf16* hr2 = hb + (rowGroup * 16 + nl) * 128;
#pragma unroll
    for (int kt = 0; kt < 4; ++kt) {
        v8bf lo = *(const v8bf*)(hr2 + kt * 32 + g * 8);
        v8bf hi = *(const v8bf*)(hr2 + kt * 32 + g * 8 + 16);
        v16bf a = __builtin_shufflevector(lo, hi, 0,1,2,3,4,5,6,7,8,9,10,11,12,13,14,15);
#pragma unroll
        for (int nt = 0; nt < 2; ++nt) {
            const int frag = (colGroup * 2 + nt) * 4 + kt;
            const v16bf bw = *(const v16bf*)(lds + 81920 + (size_t)frag * 1024 + lane * 32);
            d[nt] = __builtin_amdgcn_wmma_f32_16x16x32_bf16(false, a, false, bw, (short)0, d[nt], false, false);
        }
    }
#pragma unroll
    for (int nt = 0; nt < 2; ++nt) {
        const int col = colGroup * 32 + nt * 16 + nl;
        const float bb = b2[col];
#pragma unroll
        for (int r = 0; r < 8; ++r)
            outp[(size_t)(rowBase + g * 8 + r) * OUT_F + col] = d[nt][r] + bb;
    }
}

// ---------------------------------------------------------------------------
extern "C" void kernel_launch(void* const* d_in, const int* in_sizes, int n_in,
                              void* d_out, int out_size, void* d_ws, size_t ws_size,
                              hipStream_t stream)
{
    const float* inp     = (const float*)d_in[0];
    const float* weight  = (const float*)d_in[1];
    const float* weight2 = (const float*)d_in[2];
    const float* bias    = (const float*)d_in[3];
    const float* w1      = (const float*)d_in[4];
    const float* b1      = (const float*)d_in[5];
    const float* gamma   = (const float*)d_in[6];
    const float* beta    = (const float*)d_in[7];
    const float* w2      = (const float*)d_in[8];
    const float* b2      = (const float*)d_in[9];
    __bf16* ws = (__bf16*)d_ws;

    // Pack all weights into WMMA bf16 B-fragment order (tiny, L2-resident).
    // weight/weight2: [l][o][i] -> (l, k=i, n=o): ls=64*128, ks=1, ns=128
    pack_b_frag<<<160, 256, 0, stream>>>(weight,  ws + PW_OFF,   5, 128,  64, 64 * 128, 1, 128);
    pack_b_frag<<<160, 256, 0, stream>>>(weight2, ws + PW2_OFF,  5, 128,  64, 64 * 128, 1, 128);
    // w1: [320][128] -> (k, n): ks=128, ns=1
    pack_b_frag<<<160, 256, 0, stream>>>(w1,      ws + PW1_OFF,  1, 320, 128, 0, 128, 1);
    // w2: [128][128]
    pack_b_frag<<<64,  256, 0, stream>>>(w2,      ws + PW2M_OFF, 1, 128, 128, 0, 128, 1);

    // Stage 1: 65536 rows / 64 rows-per-block
    so3_stage1<<<1024, 256, 163840, stream>>>(inp, bias, ws + PW_OFF, ws + FEAT_OFF);
    // Stage 2: 65536 rows / 32 rows-per-block
    so3_stage2<<<2048, 256, 139264, stream>>>(ws + FEAT_OFF, ws + PW1_OFF,
                                              b1, gamma, beta, b2, (float*)d_out);
}
